// MHAttention_67748814127592
// MI455X (gfx1250) — matrix operand order repaired and checked by reading, not compiled, not run on hardware
//
#include <hip/hip_runtime.h>
#include <hip/hip_bf16.h>

// ---------------------------------------------------------------------------
// MHA forward for MI455X (gfx1250, wave32, WMMA).
//   Q = x Wq^T, K = x Wk^T, V = x Wv^T   (f16 WMMA, f32 accumulate)
//   S = Q K^T  (anti-causal mask: key < query -> -inf), softmax(S/8)
//   ctx = P V ; out = ctx Wo^T + bo      (flash-attention, no TxT tensor)
// GEMMs: 64x64 tile per wave (16 WMMA / 32-wide k-step); B tile shared by the
// 4 waves of a block through LDS, staged with async global->LDS loads
// (double-buffered, ASYNCcnt-drained).
// ---------------------------------------------------------------------------

typedef _Float16 v16h __attribute__((ext_vector_type(16)));
typedef _Float16 v8h  __attribute__((ext_vector_type(8)));
typedef _Float16 v4h  __attribute__((ext_vector_type(4)));
typedef float    v8f  __attribute__((ext_vector_type(8)));
typedef int      v4i  __attribute__((ext_vector_type(4)));

#define B_   2
#define T_   2048
#define D_   1024
#define H_   16
#define HD_  64

#if defined(__has_builtin)
#if __has_builtin(__builtin_amdgcn_global_load_async_to_lds_b128)
#define HAVE_ASYNC_LDS 1
#endif
#endif

// addrspace(1) = global ("__device__" in clang HIP diagnostics),
// addrspace(3) = LDS ("__shared__").
typedef __attribute__((address_space(1))) v4i* gas1_v4i;
typedef __attribute__((address_space(3))) v4i* as3_v4i;

__device__ __forceinline__ void wait_async0() {
#if defined(__has_builtin) && __has_builtin(__builtin_amdgcn_s_wait_asynccnt)
  __builtin_amdgcn_s_wait_asynccnt(0);
#else
  asm volatile("s_wait_asynccnt 0" ::: "memory");
#endif
}

__device__ __forceinline__ v16h cat16(v8h lo, v8h hi) {
  v16h r;
#pragma unroll
  for (int i = 0; i < 8; ++i) { r[i] = lo[i]; r[i + 8] = hi[i]; }
  return r;
}

// A fragment, 16x32 f16 (k order chosen here, same as B): lane m = lane&15, half = lane>>4,
// element j <-> k = 16*half + j  => one contiguous 32B run.
__device__ __forceinline__ v16h load_a_frag(const _Float16* __restrict__ A,
                                            int row0, int ld, int kb, int lane) {
  const int m = lane & 15, half = lane >> 4;
  const _Float16* p = A + (size_t)(row0 + m) * ld + kb + 16 * half;
  v8h lo = *(const v8h*)p;
  v8h hi = *(const v8h*)(p + 8);
  return cat16(lo, hi);
}

// B fragment, 32x16 f16 from Bt (N x K row-major): lane n = lane&15,
// element j <-> k = 16*half + j  => one contiguous 32B run.
__device__ __forceinline__ v16h load_b_frag(const _Float16* __restrict__ Bt,
                                            int col0, int ld, int kb, int lane) {
  const int n = lane & 15, half = lane >> 4;
  const _Float16* p = Bt + (size_t)(col0 + n) * ld + kb + 16 * half;
  v8h lo = *(const v8h*)p;
  v8h hi = *(const v8h*)(p + 8);
  return cat16(lo, hi);
}

__device__ __forceinline__ v8f wmma16(v16h a, v16h b, v8f c) {
  return __builtin_amdgcn_wmma_f32_16x16x32_f16(false, a, false, b,
                                                (short)0, c, false, false);
}

// ---------------------------------------------------------------------------
// fp32 -> fp16 conversion, 4 elements / thread
// ---------------------------------------------------------------------------
__global__ void __launch_bounds__(256)
cvt_f32_f16_kernel(const float* __restrict__ in, _Float16* __restrict__ out, int n4) {
  int i = blockIdx.x * blockDim.x + threadIdx.x;
  if (i < n4) {
    float4 v = ((const float4*)in)[i];
    v4h o = { (_Float16)v.x, (_Float16)v.y, (_Float16)v.z, (_Float16)v.w };
    ((v4h*)out)[i] = o;
  }
}

// ---------------------------------------------------------------------------
// GEMM: C(MxN) = A(MxK) * Bt(NxK)^T  == A @ Bt.T  (the torch Linear form).
// Block = 4 waves, block tile 256x64 (each wave 64x64). The 64-col B k-slice
// (4KB f16) is staged to LDS once per block, double-buffered.
// mode 0: out f16 scattered to (B,H,T,hd)        (Q, K)
// mode 1: out f16 scattered to (B,H,hd,T)        (V transposed)
// mode 2: out f32 row-major + bias               (final projection)
// ---------------------------------------------------------------------------
__global__ void __launch_bounds__(128)
gemm_wmma_kernel(const _Float16* __restrict__ A, const _Float16* __restrict__ Bt,
                 void* __restrict__ out, const float* __restrict__ bias,
                 int M, int N, int K, int mode) {
  __shared__ __align__(16) _Float16 btile[2][64][32];

  const int tid  = threadIdx.x;
  const int lane = tid & 31;
  const int wave = tid >> 5;
  const int half = lane >> 4;
  const int lm   = lane & 15;

  const int nbm = M >> 8;                         // 256-row block tiles
  const int m0  = (blockIdx.x % nbm) * 256 + wave * 64;
  const int n0  = (blockIdx.x / nbm) * 64;

  // Stage the 64x32 B k-slice into btile[buf]: 256 x 16B chunks, 2 per thread.
  auto stage_b = [&](int buf, int kb) {
#pragma unroll
    for (int c = tid; c < 256; c += 128) {
      const int row = c >> 2;
      const int off = (c & 3) * 8;                // in halves (16B chunks)
      const _Float16* gp = Bt + (size_t)(n0 + row) * K + kb + off;
#if defined(HAVE_ASYNC_LDS)
      __builtin_amdgcn_global_load_async_to_lds_b128(
          (gas1_v4i)gp, (as3_v4i)&btile[buf][row][off], 0, 0);
#else
      *(v8h*)&btile[buf][row][off] = *(const v8h*)gp;
#endif
    }
  };

  const v8f vzero = {0.f, 0.f, 0.f, 0.f, 0.f, 0.f, 0.f, 0.f};
  v8f acc[4][4];
#pragma unroll
  for (int i = 0; i < 4; ++i)
#pragma unroll
    for (int j = 0; j < 4; ++j) acc[i][j] = vzero;

  stage_b(0, 0);
#if defined(HAVE_ASYNC_LDS)
  wait_async0();
#endif
  __syncthreads();

  int cbuf = 0;
  for (int kb = 0; kb < K; kb += 32) {
    if (kb + 32 < K) stage_b(cbuf ^ 1, kb + 32);   // prefetch next slice

    v16h a0 = load_a_frag(A, m0,      K, kb, lane);
    v16h a1 = load_a_frag(A, m0 + 16, K, kb, lane);
    v16h a2 = load_a_frag(A, m0 + 32, K, kb, lane);
    v16h a3 = load_a_frag(A, m0 + 48, K, kb, lane);

    v16h b[4];
#pragma unroll
    for (int j = 0; j < 4; ++j) {
      const _Float16* p = &btile[cbuf][j * 16 + lm][16 * half];
      b[j] = cat16(*(const v8h*)p, *(const v8h*)(p + 8));
    }

#pragma unroll
    for (int j = 0; j < 4; ++j) {
      acc[0][j] = wmma16(a0, b[j], acc[0][j]);
      acc[1][j] = wmma16(a1, b[j], acc[1][j]);
      acc[2][j] = wmma16(a2, b[j], acc[2][j]);
      acc[3][j] = wmma16(a3, b[j], acc[3][j]);
    }

#if defined(HAVE_ASYNC_LDS)
    wait_async0();                                 // our async chunks landed
#endif
    __syncthreads();                               // everyone done with cbuf
    cbuf ^= 1;
  }

  // C layout (ISA 7.12.2): VGPR r, lane -> row = r + 8*(lane>>4), col = lane&15
#pragma unroll
  for (int i = 0; i < 4; ++i) {
#pragma unroll
    for (int j = 0; j < 4; ++j) {
#pragma unroll
      for (int r = 0; r < 8; ++r) {
        const int row = m0 + i * 16 + r + 8 * half;
        const int col = n0 + j * 16 + lm;
        const float v = acc[i][j][r];
        if (mode == 2) {
          ((float*)out)[(size_t)row * N + col] = v + bias[col];
        } else {
          const int b2 = row >> 11, t = row & (T_ - 1);
          const int h  = col >> 6,  d = col & (HD_ - 1);
          size_t addr;
          if (mode == 0) addr = (((size_t)(b2 * H_ + h) * T_ + t) * HD_ + d);
          else           addr = (((size_t)(b2 * H_ + h) * HD_ + d) * T_ + t);
          ((_Float16*)out)[addr] = (_Float16)v;
        }
      }
    }
  }
}

// ---------------------------------------------------------------------------
// Flash attention. Grid = B*H*(T/64) blocks, 128 threads (4 waves).
// Wave w owns query rows [m0, m0+16). Keys processed 32 at a time.
// Anti-causal mask (key < query masked) => loop keys from (m0 & ~31) to T;
// only the first tile can straddle the diagonal.
// ---------------------------------------------------------------------------
__global__ void __launch_bounds__(128)
flash_attn_kernel(const _Float16* __restrict__ Qh, const _Float16* __restrict__ Kh,
                  const _Float16* __restrict__ Vt, _Float16* __restrict__ Ctx) {
  __shared__ __align__(16) _Float16 lds[4][16][32];  // per-wave P tile

  const int lane = threadIdx.x & 31;
  const int wave = threadIdx.x >> 5;
  const int bh   = blockIdx.x >> 5;        // (b*H + h)
  const int qblk = blockIdx.x & 31;
  const int b    = bh >> 4, h = bh & 15;
  const int m0   = qblk * 64 + wave * 16;
  const int half = lane >> 4;
  const int lm   = lane & 15;

  const _Float16* Qb = Qh + (size_t)bh * T_ * HD_;   // (T, hd)
  const _Float16* Kb = Kh + (size_t)bh * T_ * HD_;   // (T, hd)
  const _Float16* Vb = Vt + (size_t)bh * T_ * HD_;   // (hd, T)

  const v16h aq0 = load_a_frag(Qb, m0, HD_, 0,  lane);
  const v16h aq1 = load_a_frag(Qb, m0, HD_, 32, lane);

  const v8f vzero = {0.f, 0.f, 0.f, 0.f, 0.f, 0.f, 0.f, 0.f};
  v8f o[4];
  float mrun[8], lrun[8];
#pragma unroll
  for (int c = 0; c < 4; ++c) o[c] = vzero;
#pragma unroll
  for (int r = 0; r < 8; ++r) { mrun[r] = -3.0e38f; lrun[r] = 0.0f; }

  const float scale = 0.125f;              // 1/sqrt(64)
  const int jstart = m0 & ~31;

  for (int j = jstart; j < T_; j += 32) {
    // scores S(16x32) = Q(16x64) * K_tile^T : 4 WMMAs
    v16h kb00 = load_b_frag(Kb, j,      HD_, 0,  lane);
    v16h kb01 = load_b_frag(Kb, j,      HD_, 32, lane);
    v16h kb10 = load_b_frag(Kb, j + 16, HD_, 0,  lane);
    v16h kb11 = load_b_frag(Kb, j + 16, HD_, 32, lane);
    if (j + 32 < T_) {                      // prefetch next K tile
      __builtin_prefetch(Kb + (size_t)(j + 32 + lm) * HD_, 0, 0);
    }
    v8f s0 = vzero, s1 = vzero;
    s0 = wmma16(aq0, kb00, s0);
    s0 = wmma16(aq1, kb01, s0);
    s1 = wmma16(aq0, kb10, s1);
    s1 = wmma16(aq1, kb11, s1);

    // scale + anti-causal mask (only first tile straddles diagonal)
    const bool need_mask = (j < m0 + 16);
    float p0[8], p1[8];
#pragma unroll
    for (int r = 0; r < 8; ++r) {
      float x0 = s0[r] * scale, x1 = s1[r] * scale;
      if (need_mask) {
        const int row = m0 + r + 8 * half;
        if (j + lm < row)      x0 = -__builtin_inff();
        if (j + 16 + lm < row) x1 = -__builtin_inff();
      }
      p0[r] = x0; p1[r] = x1;
    }

    // row max across the 16 lanes holding one score row
    float nm[8];
#pragma unroll
    for (int r = 0; r < 8; ++r) {
      float t = fmaxf(p0[r], p1[r]);
      t = fmaxf(t, __shfl_xor(t, 1));
      t = fmaxf(t, __shfl_xor(t, 2));
      t = fmaxf(t, __shfl_xor(t, 4));
      t = fmaxf(t, __shfl_xor(t, 8));
      nm[r] = fmaxf(mrun[r], t);
    }

    // online softmax update
#pragma unroll
    for (int r = 0; r < 8; ++r) {
      const float alpha = expf(mrun[r] - nm[r]);
      mrun[r] = nm[r];
      p0[r] = expf(p0[r] - nm[r]);
      p1[r] = expf(p1[r] - nm[r]);
      float rs = p0[r] + p1[r];
      rs += __shfl_xor(rs, 1);
      rs += __shfl_xor(rs, 2);
      rs += __shfl_xor(rs, 4);
      rs += __shfl_xor(rs, 8);
      lrun[r] = lrun[r] * alpha + rs;
#pragma unroll
      for (int c = 0; c < 4; ++c) o[c][r] *= alpha;
    }

    // C-layout -> A-fragment via per-wave LDS bounce (f16 P tile, row-major)
#pragma unroll
    for (int r = 0; r < 8; ++r) {
      lds[wave][r + 8 * half][lm]      = (_Float16)p0[r];
      lds[wave][r + 8 * half][lm + 16] = (_Float16)p1[r];
    }
    const _Float16* lp = &lds[wave][lm][16 * half];
    v16h ap = cat16(*(const v8h*)lp, *(const v8h*)(lp + 8));

    // ctx(16x64) += P(16x32) * V_tile(32x64) : 4 WMMAs, Vt rows contiguous
#pragma unroll
    for (int c = 0; c < 4; ++c) {
      const _Float16* vp = Vb + (size_t)(c * 16 + lm) * T_ + j + 16 * half;
      v16h bv = cat16(*(const v8h*)vp, *(const v8h*)(vp + 8));
      o[c] = wmma16(ap, bv, o[c]);
    }
  }

  // normalize and scatter ctx to (B, T, D) f16 row-major
#pragma unroll
  for (int r = 0; r < 8; ++r) {
    const float inv = 1.0f / lrun[r];
    const int row = m0 + r + 8 * half;
    const size_t base = ((size_t)b * T_ + row) * D_ + h * HD_ + lm;
#pragma unroll
    for (int c = 0; c < 4; ++c) {
      Ctx[base + c * 16] = (_Float16)(o[c][r] * inv);
    }
  }
}

// ---------------------------------------------------------------------------
// Host-side launch sequence (graph-capture safe: kernels only).
// Workspace layout (f16, 48 MB total):
//   xh(8MB) | Wqh Wkh Wvh Woh (2MB each) | Qh(8MB) | Kh(8MB) | Vt(8MB) | ctx(8MB)
// ---------------------------------------------------------------------------
extern "C" void kernel_launch(void* const* d_in, const int* in_sizes, int n_in,
                              void* d_out, int out_size, void* d_ws, size_t ws_size,
                              hipStream_t stream) {
  const float* x  = (const float*)d_in[0];
  const float* Wq = (const float*)d_in[1];
  const float* Wk = (const float*)d_in[2];
  const float* Wv = (const float*)d_in[3];
  const float* Wo = (const float*)d_in[4];
  const float* bo = (const float*)d_in[5];

  const size_t XN = (size_t)B_ * T_ * D_;   // 4,194,304
  const size_t WN = (size_t)D_ * D_;        // 1,048,576

  _Float16* xh  = (_Float16*)d_ws;
  _Float16* wqh = xh  + XN;
  _Float16* wkh = wqh + WN;
  _Float16* wvh = wkh + WN;
  _Float16* woh = wvh + WN;
  _Float16* qh  = woh + WN;
  _Float16* kh  = qh  + XN;
  _Float16* vt  = kh  + XN;
  _Float16* ctx = vt  + XN;

  // fp32 -> fp16 staging
  cvt_f32_f16_kernel<<<(int)(XN / 4 / 256), 256, 0, stream>>>(x,  xh,  (int)(XN / 4));
  cvt_f32_f16_kernel<<<(int)(WN / 4 / 256), 256, 0, stream>>>(Wq, wqh, (int)(WN / 4));
  cvt_f32_f16_kernel<<<(int)(WN / 4 / 256), 256, 0, stream>>>(Wk, wkh, (int)(WN / 4));
  cvt_f32_f16_kernel<<<(int)(WN / 4 / 256), 256, 0, stream>>>(Wv, wvh, (int)(WN / 4));
  cvt_f32_f16_kernel<<<(int)(WN / 4 / 256), 256, 0, stream>>>(Wo, woh, (int)(WN / 4));

  const int M = B_ * T_, N = D_, K = D_;
  const int nblocks = (M / 256) * (N / 64);      // 256x64 block tiles

  gemm_wmma_kernel<<<nblocks, 128, 0, stream>>>(xh, wqh, qh, nullptr, M, N, K, 0);
  gemm_wmma_kernel<<<nblocks, 128, 0, stream>>>(xh, wkh, kh, nullptr, M, N, K, 0);
  gemm_wmma_kernel<<<nblocks, 128, 0, stream>>>(xh, wvh, vt, nullptr, M, N, K, 1);

  flash_attn_kernel<<<B_ * H_ * (T_ / 64), 128, 0, stream>>>(qh, kh, vt, ctx);

  gemm_wmma_kernel<<<nblocks, 128, 0, stream>>>(ctx, woh, d_out, bo, M, N, K, 2);
}
